// BiLSTMTagger_22548578304232
// MI455X (gfx1250) — compile-verified, matrix operand order checked by reading, not run on hardware
//
#include <hip/hip_runtime.h>

// ---------------------------------------------------------------------------
// BiLSTM tagger for MI455X (gfx1250).  All dense GEMMs use
// v_wmma_f32_16x16x32_bf16 (bf16 inputs, f32 accumulate).  Recurrent scans are
// persistent kernels (char: 16-row workgroups with LDS-resident h + WMMA for
// h@Whh^T; sentence: one 1024-thread workgroup per direction, scalar matvec,
// since B=1 makes WMMA 1/16-occupied and the scan is latency-bound).
// ---------------------------------------------------------------------------

typedef __attribute__((ext_vector_type(16))) __bf16 v16bf;
typedef __attribute__((ext_vector_type(8)))  float  v8f;

__device__ __forceinline__ float sigf(float x) { return 1.0f / (1.0f + expf(-x)); }

// Load one WMMA A/B fragment (16 bf16 per lane) from a row-major matrix.
// p must point at  row_base + k0 + half*8   (half = lane>>4).
// Element e in 0..7  -> K = k0 + half*8 + e
// Element e in 8..15 -> K = k0 + 16 + half*8 + (e-8)        (ISA 7.12.2)
__device__ __forceinline__ v16bf load_frag(const __bf16* __restrict__ p) {
  v16bf r;
#pragma unroll
  for (int e = 0; e < 8; ++e) { r[e] = p[e]; r[8 + e] = p[16 + e]; }
  return r;
}

// ---------------------------------------------------------------------------
// Generic GEMM:  C[M,N] = act( A[M,K](bf16) * W[N,K]^T(bf16) + bias[N] )
// Each wave computes a 16x64 C strip (4 accumulators reusing one A fragment
// per K-step -> 4x A reuse, 4 back-to-back WMMAs with disjoint accumulators).
// 8 M-tiles per 256-thread block.  OOB N lanes: address-clamped only — in
// WMMA, C column n depends solely on B lane n, so a clamped lane can only
// corrupt its own column, which the store mask drops.   act: 0=id, 1=tanh.
// ---------------------------------------------------------------------------
__global__ __launch_bounds__(256) void wmma_gemm_bias(
    const __bf16* __restrict__ A, const __bf16* __restrict__ W,
    const float* __restrict__ bias, float* __restrict__ C,
    int M, int N, int K, int act) {
  const int wave = threadIdx.x >> 5;
  const int lane = threadIdx.x & 31;
  const int half = lane >> 4;
  const int lidx = lane & 15;

  const int n0 = blockIdx.x * 64;
  const int m0 = (blockIdx.y * 8 + wave) * 16;
  if (m0 >= M) return;                       // whole-wave uniform exit

  const int m = m0 + lidx;                   // A row held by this lane
  int  nrow[4];
  bool nok[4];
#pragma unroll
  for (int j = 0; j < 4; ++j) {
    const int n = n0 + j * 16 + lidx;        // W row held by this lane, tile j
    nok[j]  = (n < N);
    nrow[j] = nok[j] ? n : 0;                // clamp only; no zero-fill needed
  }

  v8f acc[4] = {};
  for (int k0 = 0; k0 < K; k0 += 32) {
    const __bf16* pa = A + (size_t)m * K + k0 + half * 8;
    __builtin_prefetch(pa + 128, 0, 1);
    const v16bf af = load_frag(pa);
#pragma unroll
    for (int j = 0; j < 4; ++j) {
      const v16bf bf = load_frag(W + (size_t)nrow[j] * K + k0 + half * 8);
      acc[j] = __builtin_amdgcn_wmma_f32_16x16x32_bf16(
          false, af, false, bf, (short)0, acc[j], false, false);
    }
  }

#pragma unroll
  for (int j = 0; j < 4; ++j) {
    if (!nok[j]) continue;
    const int   n  = n0 + j * 16 + lidx;
    const float bv = (bias != nullptr) ? bias[n] : 0.0f;
#pragma unroll
    for (int v = 0; v < 8; ++v) {
      const int row = m0 + v + (half ? 8 : 0);
      if (row < M) {
        float val = acc[j][v] + bv;
        if (act == 1) val = tanhf(val);
        C[(size_t)row * N + n] = val;
      }
    }
  }
}

// ---------------------------------------------------------------------------
// Char-level LSTM scan (one direction, one layer).  Workgroup owns 16 batch
// rows; h lives in LDS as bf16 (WMMA A-matrix), gates in LDS f32.
// xg = precomputed x@Wih^T + bih, layout (T=16, B=2048, 4H=1024).
// out layout (T, B, 512); this kernel writes columns [outOff, outOff+256).
// dir==1 applies per-row reverse_valid time indexing to both xg and out.
// ---------------------------------------------------------------------------
__global__ __launch_bounds__(256) void char_lstm_kernel(
    const float* __restrict__ xg, const float* __restrict__ bhh,
    const __bf16* __restrict__ Whh, const int* __restrict__ lengths,
    float* __restrict__ out, __bf16* __restrict__ out_bf,
    int dir, int outOff) {
  const int B = 2048, H = 256, G = 1024, T = 16, DOUT = 512;

  __shared__ __bf16 hbuf[16][256];   // 8 KB : h state, WMMA A operand
  __shared__ float  gbuf[16][1024];  // 64 KB: h @ Whh^T result

  const int tid  = threadIdx.x;
  const int wave = tid >> 5;
  const int lane = tid & 31;
  const int half = lane >> 4;
  const int lidx = lane & 15;
  const int wblk = blockIdx.x;       // which 16-row batch block

  for (int i = tid; i < 16 * 256; i += 256) ((__bf16*)hbuf)[i] = (__bf16)0.0f;

  float c[16];
#pragma unroll
  for (int j = 0; j < 16; ++j) c[j] = 0.0f;

  // pointwise mapping: thread handles (row = tid&15, h = tid>>4 + 16*j)
  const int prow = tid & 15;
  const int b    = wblk * 16 + prow;
  const int L    = lengths[b];

  __syncthreads();

  for (int t = 0; t < T; ++t) {
    // ---- gbuf = hbuf(16x256) @ Whh^T(1024x256), 8 tiles per wave ----
    v8f acc[8] = {};
    for (int k0 = 0; k0 < H; k0 += 32) {
      const v16bf af = load_frag(&hbuf[lidx][k0 + half * 8]);
#pragma unroll
      for (int j = 0; j < 8; ++j) {
        const int n = (wave * 8 + j) * 16 + lidx;
        const v16bf bf = load_frag(Whh + (size_t)n * H + k0 + half * 8);
        acc[j] = __builtin_amdgcn_wmma_f32_16x16x32_bf16(
            false, af, false, bf, (short)0, acc[j], false, false);
      }
    }
#pragma unroll
    for (int j = 0; j < 8; ++j) {
      const int n = (wave * 8 + j) * 16 + lidx;
#pragma unroll
      for (int v = 0; v < 8; ++v) gbuf[v + (half ? 8 : 0)][n] = acc[j][v];
    }
    __syncthreads();

    // ---- gate fusion + state update ----
    const int tin = (dir == 0) ? t : ((t < L) ? (L - 1 - t) : t);
    const float* xgp = xg + ((size_t)tin * B + b) * G;
#pragma unroll
    for (int j = 0; j < 16; ++j) {
      const int h = (tid >> 4) + 16 * j;
      const float gi = gbuf[prow][h]         + xgp[h]         + bhh[h];
      const float gf = gbuf[prow][H + h]     + xgp[H + h]     + bhh[H + h];
      const float gg = gbuf[prow][2 * H + h] + xgp[2 * H + h] + bhh[2 * H + h];
      const float go = gbuf[prow][3 * H + h] + xgp[3 * H + h] + bhh[3 * H + h];
      const float cn = sigf(gf) * c[j] + sigf(gi) * tanhf(gg);
      const float hn = sigf(go) * tanhf(cn);
      c[j] = cn;
      const size_t o = ((size_t)tin * B + b) * DOUT + outOff + h;
      out[o]    = hn;
      out_bf[o] = (__bf16)hn;
      hbuf[prow][h] = (__bf16)hn;
    }
    __syncthreads();
  }
}

// ---------------------------------------------------------------------------
// Attention pool: s[t,w] = a[t,w,:]·proj2 ; p = softmax_t(mask(s)) ;
// ce[w,:] = sum_t p[t] * out[t,w,:].  One 256-thread block per word.
// ---------------------------------------------------------------------------
__global__ __launch_bounds__(256) void attend_kernel(
    const float* __restrict__ a, const float* __restrict__ proj2,
    const float* __restrict__ out, const int* __restrict__ lengths,
    float* __restrict__ ce, __bf16* __restrict__ ce_bf) {
  const int B = 2048, T = 16, M = 256, D = 512;
  const int w   = blockIdx.x;
  const int tid = threadIdx.x;
  __shared__ float sv[16];
  __shared__ float p[16];

  if (tid < T) {
    const float* ap = a + ((size_t)tid * B + w) * M;
    float dot = 0.0f;
    for (int m = 0; m < M; ++m) dot += ap[m] * proj2[m];
    sv[tid] = dot;
  }
  __syncthreads();
  if (tid == 0) {
    const int L = lengths[w];
    float mx = -3.0e38f;
    for (int t = 0; t < T; ++t) {
      const float v = (t < L) ? sv[t] : -1.0e9f;
      sv[t] = v;
      mx = fmaxf(mx, v);
    }
    float s = 0.0f;
    for (int t = 0; t < T; ++t) { const float e = expf(sv[t] - mx); p[t] = e; s += e; }
    const float inv = 1.0f / s;
    for (int t = 0; t < T; ++t) p[t] *= inv;
  }
  __syncthreads();
  for (int d = tid; d < D; d += 256) {
    float v = 0.0f;
#pragma unroll
    for (int t = 0; t < T; ++t) v += p[t] * out[((size_t)t * B + w) * D + d];
    ce[(size_t)w * D + d]    = v;
    ce_bf[(size_t)w * D + d] = (__bf16)v;
  }
}

// ---------------------------------------------------------------------------
// Sentence LSTM (B=1, T=2048): one persistent 1024-thread block per direction
// (blockIdx.x = dir).  Each thread owns one gate; h state in LDS.
// xg layout (T, 1024); out layout (T, 512), fwd cols [0,256), bwd [256,512).
// ---------------------------------------------------------------------------
__global__ __launch_bounds__(1024) void sent_lstm_kernel(
    const float* __restrict__ xg_f, const float* __restrict__ xg_b,
    const float* __restrict__ WhhF, const float* __restrict__ WhhB,
    const float* __restrict__ bhhF, const float* __restrict__ bhhB,
    float* __restrict__ out, __bf16* __restrict__ out_bf, int T) {
  const int H = 256, G = 1024, DOUT = 512;
  const int dir = blockIdx.x;
  const float* xg  = dir ? xg_b : xg_f;
  const float* Whh = dir ? WhhB : WhhF;
  const float* bhh = dir ? bhhB : bhhF;

  __shared__ float hs[256];
  __shared__ float gates[1024];
  const int tid = threadIdx.x;
  if (tid < H) hs[tid] = 0.0f;
  float c = 0.0f;
  __syncthreads();

  for (int t = 0; t < T; ++t) {
    const int tin = dir ? (T - 1 - t) : t;
    const float* wr = Whh + (size_t)tid * H;
    float dot = 0.0f;
#pragma unroll 8
    for (int k = 0; k < H; ++k) dot += wr[k] * hs[k];
    gates[tid] = dot + xg[(size_t)tin * G + tid] + bhh[tid];
    __syncthreads();
    if (tid < H) {
      const float gi = gates[tid], gf = gates[H + tid];
      const float gg = gates[2 * H + tid], go = gates[3 * H + tid];
      const float cn = sigf(gf) * c + sigf(gi) * tanhf(gg);
      const float hn = sigf(go) * tanhf(cn);
      c = cn;
      const size_t o = (size_t)tin * DOUT + dir * H + tid;
      out[o]    = hn;
      out_bf[o] = (__bf16)hn;
      hs[tid]   = hn;
    }
    __syncthreads();
  }
}

// ---------------------------------------------------------------------------
// Small elementwise kernels.
// ---------------------------------------------------------------------------
__global__ void f32_to_bf16_kernel(const float* __restrict__ s,
                                   __bf16* __restrict__ d, int n) {
  const int i = blockIdx.x * blockDim.x + threadIdx.x;
  if (i < n) d[i] = (__bf16)s[i];
}

__global__ void embed_kernel(const int* __restrict__ ids,
                             const float* __restrict__ emb,
                             __bf16* __restrict__ xbf) {
  const int T = 16, B = 2048, E = 128;
  const int i = blockIdx.x * blockDim.x + threadIdx.x;
  if (i >= T * B * E) return;
  const int e = i % E;
  const int b = (i / E) % B;
  const int t = i / (E * B);
  xbf[i] = (__bf16)emb[ids[b * T + t] * E + e];
}

__global__ void log_softmax_kernel(const float* __restrict__ ts,
                                   float* __restrict__ o, int R, int C) {
  const int r = blockIdx.x * blockDim.x + threadIdx.x;
  if (r >= R) return;
  const float* p = ts + (size_t)r * C;
  float mx = p[0];
  for (int j = 1; j < C; ++j) mx = fmaxf(mx, p[j]);
  float s = 0.0f;
  for (int j = 0; j < C; ++j) s += expf(p[j] - mx);
  const float l = logf(s);
  for (int j = 0; j < C; ++j) o[(size_t)r * C + j] = p[j] - mx - l;
}

// ---------------------------------------------------------------------------
// Host orchestration.
// ---------------------------------------------------------------------------
extern "C" void kernel_launch(void* const* d_in, const int* in_sizes, int n_in,
                              void* d_out, int out_size, void* d_ws, size_t ws_size,
                              hipStream_t stream) {
  (void)in_sizes; (void)n_in; (void)out_size; (void)ws_size;
  const int T = 16, B = 2048, H = 256, E = 128, G = 1024, D = 512;
  const int TW = T * B;  // 32768

  // ---- input pointers (setup_inputs dict order, pytree-flattened) ----
  const int*   char_ids = (const int*)d_in[0];
  const int*   lengths  = (const int*)d_in[1];
  const float* emb      = (const float*)d_in[2];
  // char_params: layer{0,1} x dir{f,b} x (Wih, Whh, bih, bhh) -> d_in[3..18]
  // sent_params: same shape                                   -> d_in[19..34]
  const float* cWih[2][2], *cWhh[2][2], *cBih[2][2], *cBhh[2][2];
  const float* sWih[2][2], *sWhh[2][2], *sBih[2][2], *sBhh[2][2];
  for (int l = 0; l < 2; ++l)
    for (int dd = 0; dd < 2; ++dd) {
      int base = 3 + (l * 2 + dd) * 4;
      cWih[l][dd] = (const float*)d_in[base + 0];
      cWhh[l][dd] = (const float*)d_in[base + 1];
      cBih[l][dd] = (const float*)d_in[base + 2];
      cBhh[l][dd] = (const float*)d_in[base + 3];
      base = 19 + (l * 2 + dd) * 4;
      sWih[l][dd] = (const float*)d_in[base + 0];
      sWhh[l][dd] = (const float*)d_in[base + 1];
      sBih[l][dd] = (const float*)d_in[base + 2];
      sBhh[l][dd] = (const float*)d_in[base + 3];
    }
  const float* proj1 = (const float*)d_in[35];
  const float* proj2 = (const float*)d_in[36];
  const float* Wtag  = (const float*)d_in[37];
  const float* btag  = (const float*)d_in[38];
  float* outp = (float*)d_out;

  // ---- workspace carve-up ----
  size_t off = 0;
  auto carve = [&](size_t bytes) -> char* {
    char* p = (char*)d_ws + off;
    off = (off + bytes + 255) & ~(size_t)255;
    return p;
  };
  __bf16* x_bf    = (__bf16*)carve((size_t)TW * E * 2);
  __bf16* cWih_bf[2][2], *cWhh_bf[2][2], *sWih_bf[2][2];
  const int cKin[2] = {E, D};
  for (int l = 0; l < 2; ++l)
    for (int dd = 0; dd < 2; ++dd) {
      cWih_bf[l][dd] = (__bf16*)carve((size_t)G * cKin[l] * 2);
      cWhh_bf[l][dd] = (__bf16*)carve((size_t)G * H * 2);
      sWih_bf[l][dd] = (__bf16*)carve((size_t)G * D * 2);
    }
  __bf16* proj1_bf = (__bf16*)carve((size_t)256 * D * 2);
  __bf16* wtag_bf  = (__bf16*)carve((size_t)50 * D * 2);
  float*  xg_f     = (float*)carve((size_t)TW * G * 4);
  float*  xg_b     = (float*)carve((size_t)TW * G * 4);
  float*  out0     = (float*)carve((size_t)TW * D * 4);
  __bf16* out0_bf  = (__bf16*)carve((size_t)TW * D * 2);
  float*  out1     = (float*)carve((size_t)TW * D * 4);
  __bf16* out1_bf  = (__bf16*)carve((size_t)TW * D * 2);
  float*  a_buf    = (float*)carve((size_t)TW * 256 * 4);
  float*  ce       = (float*)carve((size_t)B * D * 4);
  __bf16* ce_bf    = (__bf16*)carve((size_t)B * D * 2);
  float*  sxg_f    = (float*)carve((size_t)B * G * 4);
  float*  sxg_b    = (float*)carve((size_t)B * G * 4);
  float*  sout0    = (float*)carve((size_t)B * D * 4);
  __bf16* sout0_bf = (__bf16*)carve((size_t)B * D * 2);
  float*  sout1    = (float*)carve((size_t)B * D * 4);
  __bf16* sout1_bf = (__bf16*)carve((size_t)B * D * 2);
  float*  tag_buf  = (float*)carve((size_t)B * 50 * 4);

  auto conv = [&](const float* s, __bf16* d, int n) {
    f32_to_bf16_kernel<<<(n + 255) / 256, 256, 0, stream>>>(s, d, n);
  };
  auto gemm = [&](const __bf16* A, const __bf16* W, const float* bias,
                  float* C, int M, int N, int K, int act) {
    dim3 g((N + 63) / 64, (M + 127) / 128);
    wmma_gemm_bias<<<g, 256, 0, stream>>>(A, W, bias, C, M, N, K, act);
  };

  // ---- weight conversions (one-time, tiny) ----
  for (int l = 0; l < 2; ++l)
    for (int dd = 0; dd < 2; ++dd) {
      conv(cWih[l][dd], cWih_bf[l][dd], G * cKin[l]);
      conv(cWhh[l][dd], cWhh_bf[l][dd], G * H);
      conv(sWih[l][dd], sWih_bf[l][dd], G * D);
    }
  conv(proj1, proj1_bf, 256 * D);
  conv(Wtag, wtag_bf, 50 * D);

  // ---- embedding gather -> bf16 (t, b, e) ----
  embed_kernel<<<(TW * E + 255) / 256, 256, 0, stream>>>(char_ids, emb, x_bf);

  // ---- char BiLSTM ----
  const __bf16* layer_in_bf = x_bf;
  float*  louts[2]    = {out0, out1};
  __bf16* louts_bf[2] = {out0_bf, out1_bf};
  for (int l = 0; l < 2; ++l) {
    const int K = cKin[l];
    gemm(layer_in_bf, cWih_bf[l][0], cBih[l][0], xg_f, TW, G, K, 0);
    gemm(layer_in_bf, cWih_bf[l][1], cBih[l][1], xg_b, TW, G, K, 0);
    char_lstm_kernel<<<B / 16, 256, 0, stream>>>(
        xg_f, cBhh[l][0], cWhh_bf[l][0], lengths, louts[l], louts_bf[l], 0, 0);
    char_lstm_kernel<<<B / 16, 256, 0, stream>>>(
        xg_b, cBhh[l][1], cWhh_bf[l][1], lengths, louts[l], louts_bf[l], 1, H);
    layer_in_bf = louts_bf[l];
  }

  // ---- attention pool ----
  gemm(out1_bf, proj1_bf, nullptr, a_buf, TW, 256, D, /*tanh*/ 1);
  attend_kernel<<<B, 256, 0, stream>>>(a_buf, proj2, out1, lengths, ce, ce_bf);

  // ---- sentence BiLSTM ----
  const __bf16* sin_bf = ce_bf;
  float*  souts[2]    = {sout0, sout1};
  __bf16* souts_bf[2] = {sout0_bf, sout1_bf};
  for (int l = 0; l < 2; ++l) {
    gemm(sin_bf, sWih_bf[l][0], sBih[l][0], sxg_f, B, G, D, 0);
    gemm(sin_bf, sWih_bf[l][1], sBih[l][1], sxg_b, B, G, D, 0);
    sent_lstm_kernel<<<2, 1024, 0, stream>>>(
        sxg_f, sxg_b, sWhh[l][0], sWhh[l][1], sBhh[l][0], sBhh[l][1],
        souts[l], souts_bf[l], B);
    sin_bf = souts_bf[l];
  }

  // ---- tag head + log_softmax ----
  gemm(sout1_bf, wtag_bf, btag, tag_buf, B, 50, D, 0);
  log_softmax_kernel<<<(B + 255) / 256, 256, 0, stream>>>(tag_buf, outp, B, 50);
}
